// MultiheadLshAttention_17274358465228
// MI455X (gfx1250) — matrix operand order repaired and checked by reading, not compiled, not run on hardware
//
#include <hip/hip_runtime.h>
#include <cstddef>

// ---------------------------------------------------------------------------
// Problem constants (from reference)
// ---------------------------------------------------------------------------
#define Tt 2048
#define Bb 2
#define Ee 1024
#define Hh 8
#define Dd 128            // E / H
#define Rr 2
#define NHASH 8
#define CH 32             // chunk length
#define Cc 64             // T / CH chunks
#define Mm 96             // 3 * CH key window

typedef __attribute__((ext_vector_type(16))) _Float16 v16h;
typedef __attribute__((ext_vector_type(8)))  _Float16 v8h;
typedef __attribute__((ext_vector_type(8)))  float    v8f;

#define SHUF16(lo, hi) __builtin_shufflevector(lo, hi, 0,1,2,3,4,5,6,7,8,9,10,11,12,13,14,15)

// ---------------------------------------------------------------------------
// Elementwise f32 -> f16 convert.
// ---------------------------------------------------------------------------
__global__ __launch_bounds__(256)
void cvt_f16_kernel(const float* __restrict__ in, _Float16* __restrict__ out, int n)
{
    const int i = blockIdx.x * 256 + threadIdx.x;
    if (i < n) out[i] = (_Float16)in[i];
}

// ---------------------------------------------------------------------------
// Convert + transpose weights: Wt[n*K + k] = (f16) W[k*N + n].
// Row-per-column layout makes WMMA B-fragment loads fully contiguous.
// ---------------------------------------------------------------------------
__global__ __launch_bounds__(256)
void cvt_t_kernel(const float* __restrict__ W, _Float16* __restrict__ Wt, int K, int N)
{
    const int idx = blockIdx.x * 256 + threadIdx.x;   // n*K + k
    if (idx >= K * N) return;
    const int k = idx % K;
    const int n = idx / K;
    Wt[idx] = (_Float16)W[(size_t)k * N + n];
}

// ---------------------------------------------------------------------------
// WMMA GEMM, f16 operands, f32 accumulate.  One wave per 16x64 output strip:
// each A fragment is reused across 4 column tiles (4 WMMAs per A load).
//   Out[Nrows,Ncols] = X[Nrows,K] @ Wt[Ncols,K]^T + bias
// ---------------------------------------------------------------------------
__device__ __forceinline__ void gemm16_core(
    const _Float16* __restrict__ X, const _Float16* __restrict__ Wt,
    int tm, int tn0, int lane, int K, v8f acc[4])
{
    const int lhalf = lane >> 4;
    const int lmod  = lane & 15;
    const _Float16* xrow = X + (size_t)(tm * 16 + lmod) * K;
    for (int k0 = 0; k0 < K; k0 += 32) {
        // A frag: two contiguous 8-half runs per lane (16B aligned)
        const v8h* ap = (const v8h*)(xrow + k0 + (lhalf << 4));
        const v16h a = SHUF16(ap[0], ap[1]);
#pragma unroll
        for (int n = 0; n < 4; ++n) {
            // B frag: one contiguous 16-half run per lane (32B aligned)
            const v16h bf = *(const v16h*)(Wt + (size_t)((tn0 + n) * 16 + lmod) * K
                                           + k0 + (lhalf ? 16 : 0));
            acc[n] = __builtin_amdgcn_wmma_f32_16x16x32_f16(false, a, false, bf,
                                                            (short)0, acc[n], false, false);
        }
    }
}

__global__ __launch_bounds__(32)
void wmma_gemm_f32out(const _Float16* __restrict__ X, const _Float16* __restrict__ Wt,
                      const float* __restrict__ bias, float* __restrict__ Out,
                      int K, int Ncols)
{
    const int tm = blockIdx.x, tn0 = blockIdx.y * 4;
    const int lane = threadIdx.x;
    const int lhalf = lane >> 4, lmod = lane & 15;
    const v8f zero = {};
    v8f acc[4]; for (int n = 0; n < 4; ++n) acc[n] = zero;
    gemm16_core(X, Wt, tm, tn0, lane, K, acc);
#pragma unroll
    for (int n = 0; n < 4; ++n) {
        const int col = (tn0 + n) * 16 + lmod;
        const float bv = bias[col];
#pragma unroll
        for (int j = 0; j < 8; ++j)
            Out[(size_t)(tm * 16 + j + lhalf * 8) * Ncols + col] = acc[n][j] + bv;
    }
}

__global__ __launch_bounds__(32)
void wmma_gemm_f16out(const _Float16* __restrict__ X, const _Float16* __restrict__ Wt,
                      const float* __restrict__ bias, _Float16* __restrict__ Out,
                      int K, int Ncols)
{
    const int tm = blockIdx.x, tn0 = blockIdx.y * 4;
    const int lane = threadIdx.x;
    const int lhalf = lane >> 4, lmod = lane & 15;
    const v8f zero = {};
    v8f acc[4]; for (int n = 0; n < 4; ++n) acc[n] = zero;
    gemm16_core(X, Wt, tm, tn0, lane, K, acc);
#pragma unroll
    for (int n = 0; n < 4; ++n) {
        const int col = (tn0 + n) * 16 + lmod;
        const float bv = bias[col];
#pragma unroll
        for (int j = 0; j < 8; ++j)
            Out[(size_t)(tm * 16 + j + lhalf * 8) * Ncols + col] = (_Float16)(acc[n][j] + bv);
    }
}

// ---------------------------------------------------------------------------
// Row L2-normalize (shared-kq keys) + query scaling; emits f16 activations.
// One 256-thread block per (t,b) row of E=1024.
// ---------------------------------------------------------------------------
__global__ __launch_bounds__(256)
void norm_scale_kernel(const float* __restrict__ qf,
                       _Float16* __restrict__ ksc, _Float16* __restrict__ qsc)
{
    __shared__ float red[256];
    const int row = blockIdx.x;                 // t*B + b
    const float* x = qf + (size_t)row * Ee;
    float ss = 0.f;
    for (int e = threadIdx.x; e < Ee; e += 256) { const float v = x[e]; ss += v * v; }
    red[threadIdx.x] = ss;
    __syncthreads();
    for (int s = 128; s > 0; s >>= 1) {
        if ((int)threadIdx.x < s) red[threadIdx.x] += red[threadIdx.x + s];
        __syncthreads();
    }
    const float rnorm   = rsqrtf(red[0]);
    const float scaling = 0.08838834764831845f; // 128^-0.5
    for (int e = threadIdx.x; e < Ee; e += 256) {
        const float v = x[e];
        ksc[(size_t)row * Ee + e] = (_Float16)(v * rnorm);
        qsc[(size_t)row * Ee + e] = (_Float16)(v * scaling);
    }
}

// ---------------------------------------------------------------------------
// LSH hashes: argmax over [lin, -lin].  One thread per (b,r,h,t).
// ---------------------------------------------------------------------------
__global__ __launch_bounds__(256)
void hash_kernel(const _Float16* __restrict__ x, const float* __restrict__ hw,
                 int* __restrict__ out)
{
    const int idx = blockIdx.x * 256 + threadIdx.x;       // ((b*R+r)*H+h)*T + t
    const int t = idx % Tt;
    const int h = (idx / Tt) % Hh;
    const int r = (idx / (Tt * Hh)) % Rr;
    const int b = idx / (Tt * Hh * Rr);
    const _Float16* xp = x + (size_t)(t * Bb + b) * Ee + h * Dd;
    const float* wp = hw + (size_t)((r * Hh + h) * Dd) * (NHASH / 2);
    float lin[NHASH / 2] = {0.f, 0.f, 0.f, 0.f};
    for (int d = 0; d < Dd; ++d) {
        const float xv = (float)xp[d];
#pragma unroll
        for (int n = 0; n < NHASH / 2; ++n) lin[n] += xv * wp[d * (NHASH / 2) + n];
    }
    int best = 0; float bv = lin[0];
#pragma unroll
    for (int n = 1; n < NHASH; ++n) {
        const float v = (n < NHASH / 2) ? lin[n] : -lin[n - NHASH / 2];
        if (v > bv) { bv = v; best = n; }   // first-max wins (matches jnp.argmax)
    }
    out[idx] = best;
}

// ---------------------------------------------------------------------------
// Stable counting sort by hash (8 bins) over T=2048 per (b,r,h).
// ---------------------------------------------------------------------------
__global__ __launch_bounds__(256)
void sort_kernel(const int* __restrict__ hash, int* __restrict__ perm,
                 int* __restrict__ inv)
{
    __shared__ int cnt[NHASH][256];
    const int g   = blockIdx.x;                 // (b,r,h)
    const int tid = threadIdx.x;
    const int* hp = hash + (size_t)g * Tt;

    int lc[NHASH];
#pragma unroll
    for (int i = 0; i < NHASH; ++i) lc[i] = 0;
    int hv[8];
    for (int j = 0; j < 8; ++j) { hv[j] = hp[tid * 8 + j]; lc[hv[j]]++; }
#pragma unroll
    for (int i = 0; i < NHASH; ++i) cnt[i][tid] = lc[i];
    __syncthreads();
    if (tid == 0) {                             // serial exclusive scan (2048 adds)
        int run = 0;
        for (int i = 0; i < NHASH; ++i)
            for (int t2 = 0; t2 < 256; ++t2) { const int tmp = cnt[i][t2]; cnt[i][t2] = run; run += tmp; }
    }
    __syncthreads();
    int off[NHASH];
#pragma unroll
    for (int i = 0; i < NHASH; ++i) off[i] = cnt[i][tid];
    int* pp = perm + (size_t)g * Tt;
    int* ip = inv ? inv + (size_t)g * Tt : nullptr;
    for (int j = 0; j < 8; ++j) {
        const int pos = off[hv[j]]++;
        pp[pos] = tid * 8 + j;
        if (ip) ip[tid * 8 + j] = pos;
    }
}

// ---------------------------------------------------------------------------
// Fused chunked LSH attention.  One 64-thread (2-wave) block per (b,r,h,c).
// f16 gathers straight into WMMA fragments; QK^T WMMA -> mask + duplicate
// penalty -> softmax (z kept) -> attn@V WMMA.
// ---------------------------------------------------------------------------
__global__ __launch_bounds__(64)
void lsh_attn_kernel(const _Float16* __restrict__ qsc, const _Float16* __restrict__ ksc,
                     const _Float16* __restrict__ vf,
                     const int* __restrict__ qh, const int* __restrict__ kh,
                     const int* __restrict__ pq, const int* __restrict__ pk,
                     const int* __restrict__ invq,
                     float* __restrict__ o_s, float* __restrict__ z_out)
{
    const int blk = blockIdx.x;
    const int c  = blk % Cc;
    const int h  = (blk / Cc) % Hh;
    const int rr = (blk / (Cc * Hh)) % Rr;
    const int b  = blk / (Cc * Hh * Rr);
    const int tid  = threadIdx.x;
    const int lane = tid & 31;
    const int wave = tid >> 5;
    const int lhalf = lane >> 4;
    const int lmod  = lane & 15;
    const int rowbase = wave * 16;

    __shared__ int      s_qid[CH];
    __shared__ int      s_kid[Mm];
    __shared__ int      s_qh[CH];
    __shared__ int      s_kh[Mm];
    __shared__ int      s_other[CH][Mm];            // other round's attended key ids
    __shared__ float    s_sc[CH][Mm];               // scores
    __shared__ __align__(32) _Float16 s_at[CH][Mm]; // attention weights (WMMA A)
    __shared__ float    s_z[CH];

    const int baseT = ((b * Rr + rr) * Hh + h) * Tt;

    for (int i = tid; i < CH; i += 64) {
        const int tq = pq[baseT + c * CH + i];
        s_qid[i] = tq;
        s_qh[i]  = qh[baseT + tq];
    }
    for (int i = tid; i < Mm; i += 64) {
        const int cc = (c + i / CH - 1 + Cc) % Cc;   // chunks {c-1,c,c+1} w/ wrap
        const int tk = pk[baseT + cc * CH + (i % CH)];
        s_kid[i] = tk;
        s_kh[i]  = kh[baseT + tk];
    }
    __syncthreads();

    // ---- stage other-round attended key ids (R=2 -> single other round) ----
    {
        const int r2 = 1 - rr;
        const int baseT2 = ((b * Rr + r2) * Hh + h) * Tt;
        for (int idx = tid; idx < CH * Mm; idx += 64) {
            const int l = idx / Mm, m2 = idx % Mm;
            const int c2 = invq[baseT2 + s_qid[l]] / CH;
            const int cc = (c2 + m2 / CH - 1 + Cc) % Cc;
            s_other[l][m2] = pk[baseT2 + cc * CH + (m2 % CH)];
        }
    }

    // ---- QK^T : 16x96 scores per wave, K = D = 128 (4 WMMA k-steps) ----
    const _Float16* qrowp = qsc + (size_t)(s_qid[rowbase + lmod] * Bb + b) * Ee + h * Dd;
    v16h afr[4];
#pragma unroll
    for (int kt = 0; kt < 4; ++kt) {
        const v8h* ap = (const v8h*)(qrowp + kt * 32 + (lhalf << 4));
        afr[kt] = SHUF16(ap[0], ap[1]);
    }
    for (int nt = 0; nt < 6; ++nt) {
        const int m = nt * 16 + lmod;
        const _Float16* krowp = ksc + (size_t)(s_kid[m] * Bb + b) * Ee + h * Dd;
        v8f acc = {};
#pragma unroll
        for (int kt = 0; kt < 4; ++kt) {
            const v16h bfr = *(const v16h*)(krowp + kt * 32 + (lhalf ? 16 : 0));
            acc = __builtin_amdgcn_wmma_f32_16x16x32_f16(false, afr[kt], false, bfr,
                                                         (short)0, acc, false, false);
        }
#pragma unroll
        for (int j = 0; j < 8; ++j)
            s_sc[rowbase + j + lhalf * 8][nt * 16 + lmod] = acc[j];
    }
    __syncthreads();

    // ---- masks + cross-round duplicate penalty ----
    for (int idx = tid; idx < CH * Mm; idx += 64) {
        const int l = idx / Mm, m = idx % Mm;
        const int kid = s_kid[m];
        float s = s_sc[l][m];
        s -= 1.0e16f * (float)(s_qh[l] != s_kh[m]);   // mask_different_hashes
        s -= 1.0e8f  * (float)(s_qid[l] == kid);      // mask_current
        int dup = 0;
        for (int m2 = 0; m2 < Mm; ++m2) dup += (kid == s_kid[m2]);       // own round
        for (int m2 = 0; m2 < Mm; ++m2) dup += (kid == s_other[l][m2]);  // other round
        s -= __logf((float)dup);
        s_sc[l][m] = s;
    }
    __syncthreads();

    // ---- row softmax + logsumexp (one thread per row) ----
    if (tid < CH) {
        const int l = tid;
        float mx = -3.0e38f;
        for (int m = 0; m < Mm; ++m) mx = fmaxf(mx, s_sc[l][m]);
        float sum = 0.f;
        for (int m = 0; m < Mm; ++m) sum += __expf(s_sc[l][m] - mx);
        const float inv = 1.f / sum;
        for (int m = 0; m < Mm; ++m) s_at[l][m] = (_Float16)(__expf(s_sc[l][m] - mx) * inv);
        s_z[l] = mx + __logf(sum);
    }
    __syncthreads();

    // ---- attn @ V : 16x128 per wave, K = M = 96 (3 WMMA k-steps) ----
    v16h atf[3];
#pragma unroll
    for (int kt = 0; kt < 3; ++kt) {
        const v8h lo = *(const v8h*)&s_at[rowbase + lmod][kt * 32 + (lhalf << 4)];
        const v8h hi = *(const v8h*)&s_at[rowbase + lmod][kt * 32 + (lhalf << 4) + 8];
        atf[kt] = SHUF16(lo, hi);
    }
    float* obase = o_s + ((size_t)baseT + c * CH) * Dd;
    for (int nt = 0; nt < 8; ++nt) {
        const int d = nt * 16 + lmod;
        v8f acc = {};
#pragma unroll
        for (int kt = 0; kt < 3; ++kt) {
            v16h bfr;
#pragma unroll
            for (int e = 0; e < 16; ++e) {
                const int m = kt * 32 + (lhalf ? 16 : 0) + e;
                bfr[e] = vf[(size_t)(s_kid[m] * Bb + b) * Ee + h * Dd + d];
            }
            acc = __builtin_amdgcn_wmma_f32_16x16x32_f16(false, atf[kt], false, bfr,
                                                         (short)0, acc, false, false);
        }
#pragma unroll
        for (int j = 0; j < 8; ++j)
            obase[(size_t)(rowbase + j + lhalf * 8) * Dd + d] = acc[j];
    }
    if (tid < CH) z_out[baseT + c * CH + tid] = s_z[tid];
}

// ---------------------------------------------------------------------------
// Un-sort outputs, combine rounds by softmax(z), emit f16 for the final GEMM.
// ---------------------------------------------------------------------------
__global__ __launch_bounds__(256)
void combine_kernel(const float* __restrict__ o_s, const float* __restrict__ z,
                    const int* __restrict__ invq, _Float16* __restrict__ acc)
{
    const size_t idx = (size_t)blockIdx.x * 256 + threadIdx.x;
    if (idx >= (size_t)Tt * Bb * Ee) return;
    const int e = (int)(idx % Ee);
    const int b = (int)((idx / Ee) % Bb);
    const int t = (int)(idx / ((size_t)Ee * Bb));
    const int h = e / Dd, d = e % Dd;
    float zs[Rr], os[Rr];
#pragma unroll
    for (int r = 0; r < Rr; ++r) {
        const int base = ((b * Rr + r) * Hh + h) * Tt;
        const int ts = invq[base + t];
        zs[r] = z[base + ts];
        os[r] = o_s[((size_t)base + ts) * Dd + d];
    }
    const float mx = fmaxf(zs[0], zs[1]);
    const float w0 = __expf(zs[0] - mx), w1 = __expf(zs[1] - mx);
    acc[idx] = (_Float16)((os[0] * w0 + os[1] * w1) / (w0 + w1));
}

// ---------------------------------------------------------------------------
// Host-side launch
// ---------------------------------------------------------------------------
extern "C" void kernel_launch(void* const* d_in, const int* in_sizes, int n_in,
                              void* d_out, int out_size, void* d_ws, size_t ws_size,
                              hipStream_t stream)
{
    (void)in_sizes; (void)n_in; (void)out_size; (void)ws_size;
    const float* query  = (const float*)d_in[0];
    /* d_in[1] (key) unused: share_kq=True */
    const float* value  = (const float*)d_in[2];
    const float* Wq     = (const float*)d_in[3];
    const float* bq     = (const float*)d_in[4];
    const float* Wv     = (const float*)d_in[5];
    const float* bv     = (const float*)d_in[6];
    const float* Wo     = (const float*)d_in[7];
    const float* bo     = (const float*)d_in[8];
    const float* hash_w = (const float*)d_in[9];
    float* out = (float*)d_out;

    char* ws = (char*)d_ws;
    size_t off = 0;
    auto alloc = [&](size_t bytes) -> void* {
        void* p = ws + off;
        off += (bytes + 255) & ~(size_t)255;
        return p;
    };
    const size_t nTBE  = (size_t)Tt * Bb * Ee;      // 4 Mi elements
    const size_t nEE   = (size_t)Ee * Ee;
    const size_t nBRHT = (size_t)Bb * Rr * Hh * Tt;

    float*     qf     = (float*)alloc(nTBE * 4);      // qf f32 (norm input)
    _Float16*  qx16   = (_Float16*)alloc(nTBE * 2);   // converted inputs
    _Float16*  vx16   = (_Float16*)alloc(nTBE * 2);
    _Float16*  WqT    = (_Float16*)alloc(nEE * 2);    // transposed f16 weights
    _Float16*  WvT    = (_Float16*)alloc(nEE * 2);
    _Float16*  WoT    = (_Float16*)alloc(nEE * 2);
    _Float16*  vf16   = (_Float16*)alloc(nTBE * 2);   // value features (f16)
    _Float16*  qsc16  = (_Float16*)alloc(nTBE * 2);   // scaled queries (f16)
    _Float16*  ksc16  = (_Float16*)alloc(nTBE * 2);   // normalized keys (f16)
    _Float16*  acc16  = (_Float16*)alloc(nTBE * 2);   // combined attention out
    float*     o_s    = (float*)alloc(nBRHT * Dd * 4);
    float*     zbuf   = (float*)alloc(nBRHT * 4);
    int*       qhB    = (int*)alloc(nBRHT * 4);
    int*       khB    = (int*)alloc(nBRHT * 4);
    int*       pqB    = (int*)alloc(nBRHT * 4);
    int*       pkB    = (int*)alloc(nBRHT * 4);
    int*       invq   = (int*)alloc(nBRHT * 4);

    // 0) one-time f16 conversion / weight transposition
    cvt_f16_kernel<<<(int)(nTBE / 256), 256, 0, stream>>>(query, qx16, (int)nTBE);
    cvt_f16_kernel<<<(int)(nTBE / 256), 256, 0, stream>>>(value, vx16, (int)nTBE);
    cvt_t_kernel<<<(int)(nEE / 256), 256, 0, stream>>>(Wq, WqT, Ee, Ee);
    cvt_t_kernel<<<(int)(nEE / 256), 256, 0, stream>>>(Wv, WvT, Ee, Ee);
    cvt_t_kernel<<<(int)(nEE / 256), 256, 0, stream>>>(Wo, WoT, Ee, Ee);

    const dim3 gemmGrid((Tt * Bb) / 16, Ee / 64);
    // 1) qf = query @ Wq + bq (f32 for normalization); vf = value @ Wv + bv (f16)
    wmma_gemm_f32out<<<gemmGrid, 32, 0, stream>>>(qx16, WqT, bq, qf, Ee, Ee);
    wmma_gemm_f16out<<<gemmGrid, 32, 0, stream>>>(vx16, WvT, bv, vf16, Ee, Ee);
    // 2) k = normalize(qf), q = qf * d^-0.5  (f16 activations)
    norm_scale_kernel<<<Tt * Bb, 256, 0, stream>>>(qf, ksc16, qsc16);
    // 3) hashes
    hash_kernel<<<(Bb * Rr * Hh * Tt) / 256, 256, 0, stream>>>(qsc16, hash_w, qhB);
    hash_kernel<<<(Bb * Rr * Hh * Tt) / 256, 256, 0, stream>>>(ksc16, hash_w, khB);
    // 4) stable sort by hash per (b,r,h)
    sort_kernel<<<Bb * Rr * Hh, 256, 0, stream>>>(qhB, pqB, invq);
    sort_kernel<<<Bb * Rr * Hh, 256, 0, stream>>>(khB, pkB, nullptr);
    // 5) fused chunked attention
    lsh_attn_kernel<<<Bb * Rr * Hh * Cc, 64, 0, stream>>>(qsc16, ksc16, vf16, qhB, khB,
                                                          pqB, pkB, invq, o_s, zbuf);
    // 6) un-sort + combine rounds (f16 out)
    combine_kernel<<<(int)((nTBE + 255) / 256), 256, 0, stream>>>(o_s, zbuf, invq, acc16);
    // 7) out = acc @ Wo + bo
    wmma_gemm_f32out<<<gemmGrid, 32, 0, stream>>>(acc16, WoT, bo, out, Ee, Ee);
}
